// DiffusionInteractionBlock_70574902608586
// MI455X (gfx1250) — matrix operand order repaired and checked by reading, not compiled, not run on hardware
//
#include <hip/hip_runtime.h>
#include <hip/hip_bf16.h>

// ---------------------------------------------------------------- constants
#define NN      10000
#define EE      160000
#define MUL     128
#define EDGE_F  8
#define MLP_IN  265
#define MLP_INP 288          // padded to multiple of 32
#define INV_SQRT_MUL 0.08838834764831845f   // 1/sqrt(128)
#define OUT_SCALE    0.00390625f            // 1/(sqrt(256)*16) = 1/256
#define INV_SQ3      0.5773502691896258f

typedef __attribute__((ext_vector_type(16))) _Float16 v16h;
typedef __attribute__((ext_vector_type(8)))  _Float16 v8h;
typedef __attribute__((ext_vector_type(8)))  float    v8f;
typedef __attribute__((ext_vector_type(4)))  int      v4i;

typedef __attribute__((address_space(1))) v4i as1_v4i;   // global
typedef __attribute__((address_space(3))) v4i as3_v4i;   // LDS

#if defined(__has_builtin)
#  if __has_builtin(__builtin_amdgcn_global_load_async_to_lds_b128)
#    define ASYNC_LDS_BUILTIN 1
#  endif
#  if __has_builtin(__builtin_amdgcn_s_wait_asynccnt)
#    define WAIT_ASYNC_BUILTIN 1
#  endif
#endif

// ---------------------------------------------------------------- helpers
__device__ __forceinline__ v16h frag2(const _Float16* p0, const _Float16* p1) {
    v8h a = *(const v8h*)p0;
    v8h b = *(const v8h*)p1;
    v16h r;
#pragma unroll
    for (int i = 0; i < 8; ++i) { r[i] = a[i]; r[i + 8] = b[i]; }
    return r;
}

__device__ __forceinline__ v8f wmma_f16(v16h a, v16h b, v8f c) {
    // D = A(16x32 f16) * B(32x16 f16) + C(16x16 f32)
    return __builtin_amdgcn_wmma_f32_16x16x32_f16(
        /*neg_a=*/false, a, /*neg_b=*/false, b,
        /*c_mod=*/(short)0, c, /*reuse_a=*/false, /*reuse_b=*/false);
}

__device__ __forceinline__ float silu_f(float x) {
    return x / (1.0f + __expf(-x));
}

// 16-byte async DMA: global -> LDS, no VGPR data path (ASYNCcnt tracked)
__device__ __forceinline__ void async_copy16(const _Float16* gsrc, _Float16* ldst) {
#if defined(ASYNC_LDS_BUILTIN)
    __builtin_amdgcn_global_load_async_to_lds_b128(
        (as1_v4i*)gsrc, (as3_v4i*)ldst, 0, 0);
#else
    unsigned int       loff = (unsigned int)(unsigned long long)(as3_v4i*)ldst;
    unsigned long long ga   = (unsigned long long)gsrc;
    asm volatile("global_load_async_to_lds_b128 %0, %1, off"
                 :: "v"(loff), "v"(ga) : "memory");
#endif
}

__device__ __forceinline__ void wait_async0() {
#if defined(WAIT_ASYNC_BUILTIN)
    __builtin_amdgcn_s_wait_asynccnt(0);
#else
    asm volatile("s_wait_asynccnt 0" ::: "memory");
#endif
}

__device__ __forceinline__ void atomic_add_f32(float* p, float v) {
    __hip_atomic_fetch_add(p, v, __ATOMIC_RELAXED, __HIP_MEMORY_SCOPE_AGENT);
}

// ---------------------------------------------------------------- prep
// Wt[n*ldk + k] = W[k*ncols + n] * scale   (zero-pad k in [K, ldk))
__global__ void transpose_f16_kernel(const float* __restrict__ W,
                                     _Float16* __restrict__ Wt,
                                     int K, int ncols, int ldk, float scale) {
    int k = blockIdx.x * blockDim.x + threadIdx.x;
    int n = blockIdx.y;
    if (k >= ldk) return;
    float v = (k < K) ? W[(size_t)k * ncols + n] * scale : 0.0f;
    Wt[(size_t)n * ldk + k] = (_Float16)v;
}

// tail16[e][0..7]=edge_feats, [8]=length, [9..31]=0  (f16)
__global__ void tail_prep_kernel(const float* __restrict__ edge_feats,
                                 const float* __restrict__ lengths,
                                 _Float16* __restrict__ tail16) {
    int idx = blockIdx.x * blockDim.x + threadIdx.x;   // EE*32 threads
    int e = idx >> 5, k = idx & 31;
    float v = (k < 8) ? edge_feats[(size_t)e * EDGE_F + k]
            : (k == 8) ? lengths[e] : 0.0f;
    tail16[(size_t)e * 32 + k] = (_Float16)v;
}

__global__ void zero_f32_kernel(float4* __restrict__ p, long long count4) {
    long long i = (long long)blockIdx.x * blockDim.x + threadIdx.x;
    long long stride = (long long)gridDim.x * blockDim.x;
    float4 z = make_float4(0.f, 0.f, 0.f, 0.f);
    for (; i < count4; i += stride) p[i] = z;
}

// ---------------------------------------------------------------- K1: node GEMMs
// y = 0: scal16 = x0 @ Wst (f16) ; y = 1: h0 = x0 @ Wu0t ; y = 2..4: h1[:,:,y-2]
__global__ void node_up_kernel(const float* __restrict__ node_feats,
                               const _Float16* __restrict__ Wst,
                               const _Float16* __restrict__ Wu0t,
                               const _Float16* __restrict__ Wu1t,
                               _Float16* __restrict__ scal16,
                               float* __restrict__ h0,
                               float* __restrict__ h1) {
    __shared__ __align__(16) _Float16 As[16][128];
    const int n0 = blockIdx.x * 16;
    const int y  = blockIdx.y;

    // branchless staging: x0 slice (stride 1) or x1 component slice (stride 3)
    const int off0 = (y < 2) ? 0 : 128 + (y - 2);
    const int kst  = (y < 2) ? 1 : 3;
    for (int idx = threadIdx.x; idx < 16 * 128; idx += 256) {
        int m = idx >> 7, k = idx & 127;
        As[m][k] = (_Float16)node_feats[(size_t)(n0 + m) * 512 + off0 + k * kst];
    }
    __syncthreads();

    const _Float16* Wt = (y == 0) ? Wst : (y == 1) ? Wu0t : Wu1t;
    const int w    = threadIdx.x >> 5;
    const int lane = threadIdx.x & 31;
    const int l15  = lane & 15;
    const int hi   = lane >> 4;
    const int colc = w * 16 + l15;

    v8f c = {};
#pragma unroll
    for (int ks = 0; ks < 128; ks += 32) {
        v16h a = frag2(&As[l15][ks + hi * 8], &As[l15][ks + 16 + hi * 8]);
        const _Float16* bp = &Wt[(size_t)colc * 128 + ks + hi * 8];
        v16h b = frag2(bp, bp + 16);
        c = wmma_f16(a, b, c);
    }

    if (y == 0) {
#pragma unroll
        for (int r = 0; r < 8; ++r) {
            size_t row = (size_t)(n0 + r + 8 * hi);
            scal16[row * 128 + colc] = (_Float16)c[r];
        }
    } else {
        float* dst = (y == 1) ? h0 : h1 + (y - 2);
        const int es = (y == 1) ? 1 : 3;
#pragma unroll
        for (int r = 0; r < 8; ++r) {
            size_t row = (size_t)(n0 + r + 8 * hi);
            dst[(row * 128 + colc) * es] = c[r];
        }
    }
}

// ---------------------------------------------------------------- K2: fused edge MLP + messages + scatter
__global__ void edge_kernel(const _Float16* __restrict__ scal16,  // [N][128]
                            const _Float16* __restrict__ tail16,  // [E][32]
                            const float* __restrict__ h0,
                            const float* __restrict__ h1,
                            const float* __restrict__ edge_attrs,
                            const int*   __restrict__ edge_index,
                            const float* __restrict__ b1,
                            const float* __restrict__ b2,
                            const _Float16* __restrict__ W1t,   // [128][288]
                            const _Float16* __restrict__ W2t,   // [128][128]
                            const _Float16* __restrict__ W3t,   // [512][128]
                            float* __restrict__ agg0,           // [N][256]
                            float* __restrict__ agg1) {         // [N][256][3]
    __shared__ __align__(16) _Float16 in_s[16][MLP_INP];
    __shared__ __align__(16) _Float16 h_s[16][128];
    __shared__ int   s_snd[16];
    __shared__ int   s_rcv[16];
    __shared__ float s_y0[16];
    __shared__ float s_y1[16][3];

    const int e0 = blockIdx.x * 16;
    const int tid = threadIdx.x;

    if (tid < 16) {
        int e = e0 + tid;
        s_snd[tid] = edge_index[e];
        s_rcv[tid] = edge_index[EE + e];
        s_y0[tid]  = edge_attrs[(size_t)e * 4 + 0];
        s_y1[tid][0] = edge_attrs[(size_t)e * 4 + 1];
        s_y1[tid][1] = edge_attrs[(size_t)e * 4 + 2];
        s_y1[tid][2] = edge_attrs[(size_t)e * 4 + 3];
    }
    __syncthreads();

    // stage MLP input via async global->LDS DMA, 16B chunks, no VGPR data:
    // row m (288 halves) = [scal16[snd] (16 chunks) | scal16[rcv] (16) | tail16[e] (4)]
    for (int idx = tid; idx < 16 * 36; idx += 256) {
        int m = idx / 36;
        int p = idx - m * 36;
        const _Float16* src;
        if (p < 16)      src = &scal16[(size_t)s_snd[m] * 128 + p * 8];
        else if (p < 32) src = &scal16[(size_t)s_rcv[m] * 128 + (p - 16) * 8];
        else             src = &tail16[(size_t)(e0 + m) * 32 + (p - 32) * 8];
        async_copy16(src, &in_s[m][p * 8]);
    }
    wait_async0();
    __syncthreads();

    const int w    = tid >> 5;
    const int lane = tid & 31;
    const int l15  = lane & 15;
    const int hi   = lane >> 4;
    const int colc = w * 16 + l15;          // output channel of this lane

    // ---- stage 1: h = silu(in @ W1 + b1) ----
    {
        float bias = b1[colc];
        v8f c = {};
#pragma unroll
        for (int ks = 0; ks < MLP_INP; ks += 32) {
            v16h a = frag2(&in_s[l15][ks + hi * 8], &in_s[l15][ks + 16 + hi * 8]);
            const _Float16* bp = &W1t[(size_t)colc * MLP_INP + ks + hi * 8];
            v16h b = frag2(bp, bp + 16);
            c = wmma_f16(a, b, c);
        }
#pragma unroll
        for (int r = 0; r < 8; ++r)
            h_s[r + 8 * hi][colc] = (_Float16)silu_f(c[r] + bias);
    }
    __syncthreads();

    // ---- stage 2: h2 = silu(h @ W2 + b2), reuse in_s storage ----
    _Float16* h2f = &in_s[0][0];            // [16][128]
    {
        float bias = b2[colc];
        v8f c = {};
#pragma unroll
        for (int ks = 0; ks < 128; ks += 32) {
            v16h a = frag2(&h_s[l15][ks + hi * 8], &h_s[l15][ks + 16 + hi * 8]);
            const _Float16* bp = &W2t[(size_t)colc * 128 + ks + hi * 8];
            v16h b = frag2(bp, bp + 16);
            c = wmma_f16(a, b, c);
        }
#pragma unroll
        for (int r = 0; r < 8; ++r)
            h2f[(r + 8 * hi) * 128 + colc] = (_Float16)silu_f(c[r] + bias);
    }
    __syncthreads();

    // ---- stage 3: tpw = h2 @ W3, 4 regions (wA,wD,wB,wC) at same channel ----
    v8f cA = {}, cD = {}, cB = {}, cC = {};
#pragma unroll
    for (int ks = 0; ks < 128; ks += 32) {
        v16h a = frag2(&h2f[l15 * 128 + ks + hi * 8],
                       &h2f[l15 * 128 + ks + 16 + hi * 8]);
        const _Float16* bpA = &W3t[(size_t)(0 * 128 + colc) * 128 + ks + hi * 8];
        const _Float16* bpD = &W3t[(size_t)(1 * 128 + colc) * 128 + ks + hi * 8];
        const _Float16* bpB = &W3t[(size_t)(2 * 128 + colc) * 128 + ks + hi * 8];
        const _Float16* bpC = &W3t[(size_t)(3 * 128 + colc) * 128 + ks + hi * 8];
        cA = wmma_f16(a, frag2(bpA, bpA + 16), cA);
        cD = wmma_f16(a, frag2(bpD, bpD + 16), cD);
        cB = wmma_f16(a, frag2(bpB, bpB + 16), cB);
        cC = wmma_f16(a, frag2(bpC, bpC + 16), cC);
    }

    // ---- messages + scatter-add (native f32 atomics, agent scope) ----
#pragma unroll
    for (int r = 0; r < 8; ++r) {
        int m  = r + 8 * hi;
        int s  = s_snd[m];
        int rc = s_rcv[m];
        float y0  = s_y0[m];
        float y1x = s_y1[m][0], y1y = s_y1[m][1], y1z = s_y1[m][2];

        float x0v = h0[(size_t)s * 128 + colc];
        const float* x1p = &h1[((size_t)s * 128 + colc) * 3];
        float x1x = x1p[0], x1y = x1p[1], x1z = x1p[2];

        float wA = cA[r], wD = cD[r], wB = cB[r], wC = cC[r];

        atomic_add_f32(&agg0[(size_t)rc * 256 + colc], x0v * y0 * wA);
        float dot = (x1x * y1x + x1y * y1y + x1z * y1z) * INV_SQ3;
        atomic_add_f32(&agg0[(size_t)rc * 256 + 128 + colc], dot * wD);

        float* pB = &agg1[((size_t)rc * 256 + colc) * 3];
        atomic_add_f32(&pB[0], x0v * y1x * wB);
        atomic_add_f32(&pB[1], x0v * y1y * wB);
        atomic_add_f32(&pB[2], x0v * y1z * wB);

        float* pC = &agg1[((size_t)rc * 256 + 128 + colc) * 3];
        atomic_add_f32(&pC[0], x1x * y0 * wC);
        atomic_add_f32(&pC[1], x1y * y0 * wC);
        atomic_add_f32(&pC[2], x1z * y0 * wC);
    }
}

// ---------------------------------------------------------------- K3: output GEMMs
// j = 0: out[:,:,0] = agg0 @ Wout0*scale ; j = 1..3: out[:,:,j] = agg1[:,:,j-1] @ Wout1*scale
__global__ void out_kernel(const float* __restrict__ agg0,
                           const float* __restrict__ agg1,
                           const _Float16* __restrict__ Wo0t,   // [128][256]
                           const _Float16* __restrict__ Wo1t,   // [128][256]
                           float* __restrict__ out) {           // [N][128][4]
    __shared__ __align__(16) _Float16 As[16][256];
    const int n0 = blockIdx.x * 16;
    const int j  = blockIdx.y;

    const float* src = (j == 0) ? agg0 : agg1 + (j - 1);
    const int es = (j == 0) ? 1 : 3;
    for (int idx = threadIdx.x; idx < 16 * 256; idx += 256) {
        int m = idx >> 8, k = idx & 255;
        As[m][k] = (_Float16)src[((size_t)(n0 + m) * 256 + k) * es];
    }
    __syncthreads();

    const _Float16* Wt = (j == 0) ? Wo0t : Wo1t;
    const int w    = threadIdx.x >> 5;
    const int lane = threadIdx.x & 31;
    const int l15  = lane & 15;
    const int hi   = lane >> 4;
    const int colc = w * 16 + l15;

    v8f c = {};
#pragma unroll
    for (int ks = 0; ks < 256; ks += 32) {
        v16h a = frag2(&As[l15][ks + hi * 8], &As[l15][ks + 16 + hi * 8]);
        const _Float16* bp = &Wt[(size_t)colc * 256 + ks + hi * 8];
        v16h b = frag2(bp, bp + 16);
        c = wmma_f16(a, b, c);
    }

#pragma unroll
    for (int r = 0; r < 8; ++r) {
        int m = r + 8 * hi;
        out[(((size_t)(n0 + m)) * 128 + colc) * 4 + j] = c[r];
    }
}

// ---------------------------------------------------------------- launch
extern "C" void kernel_launch(void* const* d_in, const int* in_sizes, int n_in,
                              void* d_out, int out_size, void* d_ws, size_t ws_size,
                              hipStream_t stream) {
    const float* node_feats = (const float*)d_in[0];
    const float* edge_attrs = (const float*)d_in[1];
    const float* edge_feats = (const float*)d_in[2];
    const float* lengths    = (const float*)d_in[3];
    const int*   edge_index = (const int*)d_in[4];
    const float* W_scalar   = (const float*)d_in[5];
    const float* W_up0      = (const float*)d_in[6];
    const float* W_up1      = (const float*)d_in[7];
    const float* W1         = (const float*)d_in[8];
    const float* b1         = (const float*)d_in[9];
    const float* W2         = (const float*)d_in[10];
    const float* b2         = (const float*)d_in[11];
    const float* W3         = (const float*)d_in[12];
    const float* Wout0      = (const float*)d_in[13];
    const float* Wout1      = (const float*)d_in[14];
    float* out = (float*)d_out;

    // ---- workspace layout (bytes) ----
    char* ws = (char*)d_ws;
    float* h0   = (float*)(ws);                    // N*128        =  5,120,000 B
    float* h1   = (float*)(ws +  5120000);         // N*128*3      = 15,360,000 B
    float* agg0 = (float*)(ws + 20480000);         // N*256        = 10,240,000 B
    float* agg1 = (float*)(ws + 30720000);         // N*256*3      = 30,720,000 B
    _Float16* hbase = (_Float16*)(ws + 61440000);  // f16 region
    _Float16* Wst    = hbase;                      // 128*128
    _Float16* Wu0t   = hbase + 16384;              // 128*128
    _Float16* Wu1t   = hbase + 32768;              // 128*128
    _Float16* W1t    = hbase + 49152;              // 128*288
    _Float16* W2t    = hbase + 86016;              // 128*128
    _Float16* W3t    = hbase + 102400;             // 512*128
    _Float16* Wo0t   = hbase + 167936;             // 128*256
    _Float16* Wo1t   = hbase + 200704;             // 128*256
    _Float16* scal16 = hbase + 233472;             // N*128
    _Float16* tail16 = hbase + 1513472;            // E*32

    // ---- weight prep (transpose -> f16, fold scales) ----
    auto tp = [&](const float* W, _Float16* Wt, int K, int ncols, int ldk, float scale) {
        dim3 g((ldk + 255) / 256, ncols);
        transpose_f16_kernel<<<g, 256, 0, stream>>>(W, Wt, K, ncols, ldk, scale);
    };
    tp(W_scalar, Wst,  128, 128, 128, INV_SQRT_MUL);
    tp(W_up0,    Wu0t, 128, 128, 128, INV_SQRT_MUL);
    tp(W_up1,    Wu1t, 128, 128, 128, INV_SQRT_MUL);
    tp(W1,       W1t,  MLP_IN, 128, MLP_INP, 1.0f);
    tp(W2,       W2t,  128, 128, 128, 1.0f);
    tp(W3,       W3t,  128, 512, 128, 1.0f);
    tp(Wout0,    Wo0t, 256, 128, 256, OUT_SCALE);
    tp(Wout1,    Wo1t, 256, 128, 256, OUT_SCALE);

    // ---- per-edge f16 tail [edge_feats | length | 0-pad] ----
    tail_prep_kernel<<<(EE * 32) / 256, 256, 0, stream>>>(edge_feats, lengths, tail16);

    // ---- zero scatter buffers (agg0 + agg1 contiguous: N*1024 floats) ----
    zero_f32_kernel<<<2048, 256, 0, stream>>>((float4*)agg0, (long long)NN * 1024 / 4);

    // ---- node GEMMs ----
    node_up_kernel<<<dim3(NN / 16, 5), 256, 0, stream>>>(
        node_feats, Wst, Wu0t, Wu1t, scal16, h0, h1);

    // ---- fused edge MLP + tensor-product messages + scatter ----
    edge_kernel<<<EE / 16, 256, 0, stream>>>(
        scal16, tail16, h0, h1, edge_attrs, edge_index,
        b1, b2, W1t, W2t, W3t, agg0, agg1);

    // ---- output GEMMs ----
    out_kernel<<<dim3(NN / 16, 4), 256, 0, stream>>>(agg0, agg1, Wo0t, Wo1t, out);
}
